// Net_41188736369461
// MI455X (gfx1250) — matrix operand order repaired and checked by reading, not compiled, not run on hardware
//
#include <hip/hip_runtime.h>

typedef __attribute__((ext_vector_type(16))) __bf16 v16bf;
typedef __attribute__((ext_vector_type(8)))  float  v8f;

#define BB 256
#define TT 512
#define II 64
#define HH 256
#define GG 1024      // 4*H
#define OO 4
#define D5H 1280     // 5*H
#define NWG 64       // persistent grid size
#define NTH 128      // 4 waves of 32

// ---------- helpers ----------
__device__ inline unsigned short f2bf(float f) {
  unsigned u = __float_as_uint(f);
  unsigned r = u + 0x7FFFu + ((u >> 16) & 1u);   // round-to-nearest-even
  return (unsigned short)(r >> 16);
}
__device__ inline float sigf(float x) { return 1.0f / (1.0f + __expf(-x)); }
__device__ inline float tanh_fast(float x) { return 2.0f * sigf(2.0f * x) - 1.0f; }

union FragBF { v16bf v; uint4 q[2]; };

// ---------- prep kernels ----------
__global__ void k_f32_to_bf16(const float* __restrict__ src,
                              unsigned short* __restrict__ dst, int n) {
  int i = blockIdx.x * blockDim.x + threadIdx.x;
  if (i < n) dst[i] = f2bf(src[i]);
}

__global__ void k_zero_u32(unsigned* __restrict__ p, int nwords) {
  int i = blockIdx.x * blockDim.x + threadIdx.x;
  if (i < nwords) p[i] = 0u;
}

__global__ void k_bias_sum(const float* __restrict__ a, const float* __restrict__ b,
                           float* __restrict__ out, int n) {
  int i = blockIdx.x * blockDim.x + threadIdx.x;
  if (i < n) out[i] = a[i] + b[i];
}

// Pack combined [Wih | Whh] into WMMA B-fragment order, bf16:
// packed[((nt*KC + kc)*32 + lane)*16 + e] = Wcat[n = nt*16 + (lane&15)][k = kc*32 + (e>>3)*16 + (lane>>4)*8 + (e&7)]
__global__ void k_pack_w(const float* __restrict__ Wih, const float* __restrict__ Whh,
                         unsigned short* __restrict__ packed, int Ilen, int KC) {
  int idx = blockIdx.x * blockDim.x + threadIdx.x;
  int total = 64 * KC * 32;
  if (idx >= total) return;
  int nt   = idx / (KC * 32);
  int rem  = idx % (KC * 32);
  int kc   = rem / 32;
  int lane = rem % 32;
  int lo = lane & 15, hi = lane >> 4;
  int n = nt * 16 + lo;
  unsigned short* out = packed + (size_t)idx * 16;
  for (int e = 0; e < 16; ++e) {
    int k = kc * 32 + (e >> 3) * 16 + hi * 8 + (e & 7);
    float v = (k < Ilen) ? Wih[(size_t)n * Ilen + k]
                         : Whh[(size_t)n * HH + (k - Ilen)];
    out[e] = f2bf(v);
  }
}

// ---------- grid-wide sense barrier ----------
__device__ inline void grid_sync(unsigned* bar) {
  __threadfence();
  __syncthreads();
  if (threadIdx.x == 0) {
    volatile unsigned* gen = bar + 1;
    unsigned g = *gen;
    if (atomicAdd(bar, 1u) == NWG - 1u) {
      bar[0] = 0u;
      __threadfence();
      atomicAdd((unsigned*)gen, 1u);
    } else {
      while (*gen == g) { __builtin_amdgcn_s_sleep(1); }
    }
  }
  __syncthreads();
  __threadfence();
}

// ---------- persistent LSTM layer ----------
// Grid: 64 WGs = (mb 0..3 batch-block of 64 rows) x (jg 0..15 hidden-col group of 16).
// Block: 128 threads = 4 waves; wave w handles batch rows mb*64 + w*16 .. +15, cols jg*16..+15,
// all 4 gates, accumulating in f32. Weights live in LDS for all T steps.
__global__ __launch_bounds__(NTH, 1) void k_lstm_layer(
    const unsigned short* __restrict__ xsrc, int xstrideB, int xstrideT, int Ilen,
    const unsigned short* __restrict__ wpacked, const float* __restrict__ bc,
    unsigned short* __restrict__ hbuf0, unsigned short* __restrict__ hbuf1,
    unsigned short* __restrict__ hseq_out,   // bf16 [t][b][h] or nullptr
    float* __restrict__ hlast_out,           // f32 [b][h] or nullptr
    unsigned* __restrict__ bar, int K, int KC)
{
  extern __shared__ char smem[];
  const int pstride = K * 2 + 32;                 // padded A-panel row stride (bytes)
  unsigned short* wlds = (unsigned short*)smem;   // 4 gates * KC * 32 lanes * 32B
  char* panel = smem + (size_t)4 * KC * 1024;     // 64 rows * pstride

  const int mb   = blockIdx.x & 3;
  const int jg   = blockIdx.x >> 2;
  const int lane = threadIdx.x & 31;
  const int wave = threadIdx.x >> 5;
  const int lo = lane & 15, hi = lane >> 4;
  const int j0   = jg * 16;
  const int mrow = mb * 64 + wave * 16;
  const size_t gstep = (size_t)KC * 1024;         // bytes per gate slab in LDS

  // ---- copy this WG's weight slabs (4 gate ntiles) into LDS once ----
  {
    const uint4* src = (const uint4*)wpacked;
    uint4* dst = (uint4*)wlds;
    const int cpg = KC * 64;                      // uint4 per ntile
    for (int g = 0; g < 4; ++g) {
      const uint4* s = src + (size_t)(g * 16 + jg) * cpg;
      uint4* d = dst + (size_t)g * cpg;
      for (int c = threadIdx.x; c < cpg; c += NTH) d[c] = s[c];
    }
  }
  __syncthreads();

  const float bias0 = bc[0 * HH + j0 + lo];
  const float bias1 = bc[1 * HH + j0 + lo];
  const float bias2 = bc[2 * HH + j0 + lo];
  const float bias3 = bc[3 * HH + j0 + lo];

  v8f cstate;
  for (int r = 0; r < 8; ++r) cstate[r] = 0.0f;

  const int CPR  = K / 8;      // 16B chunks per panel row
  const int CPRX = Ilen / 8;   // chunks coming from x

  for (int t = 0; t < TT; ++t) {
    const unsigned short* hprev = (t & 1) ? hbuf0 : hbuf1;  // t=0 reads zeroed hbuf1
    unsigned short*       hcur  = (t & 1) ? hbuf1 : hbuf0;

    // ---- assemble A-panel [x_t | h_{t-1}] for our 64 batch rows ----
    for (int idx = threadIdx.x; idx < 64 * CPR; idx += NTH) {
      int m = idx / CPR, c = idx % CPR;
      int brow = mb * 64 + m;
      uint4 v;
      if (c < CPRX)
        v = *(const uint4*)((const char*)xsrc +
              ((size_t)brow * xstrideB + (size_t)t * xstrideT) * 2 + (size_t)c * 16);
      else
        v = *(const uint4*)((const char*)hprev +
              (size_t)brow * HH * 2 + (size_t)(c - CPRX) * 16);
      *(uint4*)(panel + (size_t)m * pstride + (size_t)c * 16) = v;
    }
    __syncthreads();

    // ---- GEMM: 4 gate accumulators, K swept in chunks of 32 ----
    v8f a0, a1, a2, a3;
    for (int r = 0; r < 8; ++r) { a0[r] = bias0; a1[r] = bias1; a2[r] = bias2; a3[r] = bias3; }

    const char* arow = panel + (size_t)(wave * 16 + lo) * pstride + hi * 16;
    for (int kc = 0; kc < KC; ++kc) {
      FragBF a;
      a.q[0] = *(const uint4*)(arow + kc * 64);        // K = kc*32 + hi*8 + 0..7
      a.q[1] = *(const uint4*)(arow + kc * 64 + 32);   // K = kc*32 + 16 + hi*8 + 0..7
      const char* bb = (const char*)wlds + (size_t)kc * 1024 + (size_t)lane * 32;
      FragBF b;
      b.q[0] = *(const uint4*)(bb);             b.q[1] = *(const uint4*)(bb + 16);
      a0 = __builtin_amdgcn_wmma_f32_16x16x32_bf16(false, a.v, false, b.v, (short)0, a0, false, false);
      b.q[0] = *(const uint4*)(bb + gstep);     b.q[1] = *(const uint4*)(bb + gstep + 16);
      a1 = __builtin_amdgcn_wmma_f32_16x16x32_bf16(false, a.v, false, b.v, (short)0, a1, false, false);
      b.q[0] = *(const uint4*)(bb + 2 * gstep); b.q[1] = *(const uint4*)(bb + 2 * gstep + 16);
      a2 = __builtin_amdgcn_wmma_f32_16x16x32_bf16(false, a.v, false, b.v, (short)0, a2, false, false);
      b.q[0] = *(const uint4*)(bb + 3 * gstep); b.q[1] = *(const uint4*)(bb + 3 * gstep + 16);
      a3 = __builtin_amdgcn_wmma_f32_16x16x32_bf16(false, a.v, false, b.v, (short)0, a3, false, false);
    }

    // ---- lane-local gate fusion: c,h update (C layout: elem r -> M = hi*8+r, N = lo) ----
    for (int r = 0; r < 8; ++r) {
      float ig = sigf(a0[r]);
      float fg = sigf(a1[r]);
      float gg = tanh_fast(a2[r]);
      float og = sigf(a3[r]);
      float cn = fg * cstate[r] + ig * gg;
      cstate[r] = cn;
      float hv = og * tanh_fast(cn);
      int m = mrow + hi * 8 + r;
      int j = j0 + lo;
      unsigned short hb = f2bf(hv);
      hcur[(size_t)m * HH + j] = hb;
      if (hseq_out)  hseq_out[(size_t)t * (BB * HH) + (size_t)m * HH + j] = hb;
      if (hlast_out && t == TT - 1) hlast_out[(size_t)m * HH + j] = hv;
    }

    grid_sync(bar);   // h_t visible to all WGs before next panel fill
  }
}

// ---------- dense tail ----------
__global__ void k_dense1(const float* __restrict__ last, const float* __restrict__ W1,
                         const float* __restrict__ b1, float* __restrict__ y1) {
  int idx = blockIdx.x * blockDim.x + threadIdx.x;
  if (idx >= BB * D5H) return;
  int b = idx / D5H, n = idx % D5H;
  const float* lr = last + (size_t)b * HH;
  const float* wr = W1 + (size_t)n * HH;
  float s = b1[n];
  for (int k = 0; k < HH; ++k) s += lr[k] * wr[k];
  y1[idx] = s;
}

__global__ void k_dense2(const float* __restrict__ y1, const float* __restrict__ W2,
                         const float* __restrict__ b2, float* __restrict__ out) {
  int idx = blockIdx.x * blockDim.x + threadIdx.x;
  if (idx >= BB * OO) return;
  int b = idx / OO, o = idx % OO;
  const float* yr = y1 + (size_t)b * D5H;
  const float* wr = W2 + (size_t)o * D5H;
  float s = b2[o];
  for (int k = 0; k < D5H; ++k) s += yr[k] * wr[k];
  out[idx] = s;
}

// ---------- launch ----------
static inline size_t alignup(size_t x) { return (x + 255) & ~(size_t)255; }

extern "C" void kernel_launch(void* const* d_in, const int* in_sizes, int n_in,
                              void* d_out, int out_size, void* d_ws, size_t ws_size,
                              hipStream_t stream) {
  (void)in_sizes; (void)n_in; (void)out_size; (void)ws_size;
  const float* x    = (const float*)d_in[0];
  const float* Wih0 = (const float*)d_in[1];
  const float* Whh0 = (const float*)d_in[2];
  const float* bih0 = (const float*)d_in[3];
  const float* bhh0 = (const float*)d_in[4];
  const float* Wih1 = (const float*)d_in[5];
  const float* Whh1 = (const float*)d_in[6];
  const float* bih1 = (const float*)d_in[7];
  const float* bhh1 = (const float*)d_in[8];
  const float* W1   = (const float*)d_in[9];
  const float* b1   = (const float*)d_in[10];
  const float* W2   = (const float*)d_in[11];
  const float* b2   = (const float*)d_in[12];
  float* out = (float*)d_out;

  const int K0 = II + HH, KC0 = K0 / 32;   // 320, 10
  const int K1 = HH + HH, KC1 = K1 / 32;   // 512, 16

  // workspace carve-up
  char* ws = (char*)d_ws;
  size_t off = 0;
  unsigned* bar = (unsigned*)(ws + off);                  off = alignup(off + 256);
  unsigned short* xbf    = (unsigned short*)(ws + off);   off = alignup(off + (size_t)BB * TT * II * 2);
  unsigned short* h1seq  = (unsigned short*)(ws + off);   off = alignup(off + (size_t)TT * BB * HH * 2);
  unsigned short* hbuf0  = (unsigned short*)(ws + off);   off = alignup(off + (size_t)BB * HH * 2);
  unsigned short* hbuf1  = (unsigned short*)(ws + off);   off = alignup(off + (size_t)BB * HH * 2);
  unsigned short* w0p    = (unsigned short*)(ws + off);   off = alignup(off + (size_t)64 * KC0 * 512 * 2);
  unsigned short* w1p    = (unsigned short*)(ws + off);   off = alignup(off + (size_t)64 * KC1 * 512 * 2);
  float* bc0    = (float*)(ws + off);                     off = alignup(off + (size_t)GG * 4);
  float* bc1    = (float*)(ws + off);                     off = alignup(off + (size_t)GG * 4);
  float* h2last = (float*)(ws + off);                     off = alignup(off + (size_t)BB * HH * 4);
  float* y1buf  = (float*)(ws + off);                     off = alignup(off + (size_t)BB * D5H * 4);

  // prep
  {
    int n = BB * TT * II;
    k_f32_to_bf16<<<(n + 255) / 256, 256, 0, stream>>>(x, xbf, n);
    int t0 = 64 * KC0 * 32, t1 = 64 * KC1 * 32;
    k_pack_w<<<(t0 + 255) / 256, 256, 0, stream>>>(Wih0, Whh0, w0p, II, KC0);
    k_pack_w<<<(t1 + 255) / 256, 256, 0, stream>>>(Wih1, Whh1, w1p, HH, KC1);
    k_bias_sum<<<(GG + 255) / 256, 256, 0, stream>>>(bih0, bhh0, bc0, GG);
    k_bias_sum<<<(GG + 255) / 256, 256, 0, stream>>>(bih1, bhh1, bc1, GG);
  }

  const int hwords = (BB * HH * 2 * 2) / 4;   // both ping-pong buffers (contiguous)
  const size_t smem0 = (size_t)4 * KC0 * 1024 + (size_t)64 * (K0 * 2 + 32); // 40960 + 43008
  const size_t smem1 = (size_t)4 * KC1 * 1024 + (size_t)64 * (K1 * 2 + 32); // 65536 + 67584

  // layer 0: x (bf16, [b][t][i]) -> h1seq
  k_zero_u32<<<1, 64, 0, stream>>>(bar, 64);
  k_zero_u32<<<(hwords + 255) / 256, 256, 0, stream>>>((unsigned*)hbuf0, hwords);
  k_lstm_layer<<<NWG, NTH, smem0, stream>>>(
      xbf, TT * II, II, II, w0p, bc0, hbuf0, hbuf1,
      h1seq, (float*)nullptr, bar, K0, KC0);

  // layer 1: h1seq ([t][b][h]) -> h2last
  k_zero_u32<<<1, 64, 0, stream>>>(bar, 64);
  k_zero_u32<<<(hwords + 255) / 256, 256, 0, stream>>>((unsigned*)hbuf0, hwords);
  k_lstm_layer<<<NWG, NTH, smem1, stream>>>(
      h1seq, HH, BB * HH, HH, w1p, bc1, hbuf0, hbuf1,
      (unsigned short*)nullptr, h2last, bar, K1, KC1);

  // dense tail
  k_dense1<<<(BB * D5H + 255) / 256, 256, 0, stream>>>(h2last, W1, b1, y1buf);
  k_dense2<<<(BB * OO + 255) / 256, 256, 0, stream>>>(y1buf, W2, b2, out);
}
